// GRU_ODE_Delay_36584531427600
// MI455X (gfx1250) — matrix operand order repaired and checked by reading, not compiled, not run on hardware
//
#include <hip/hip_runtime.h>

// ---------------------------------------------------------------------------
// GRU-ODE step on MI455X (gfx1250, wave32, WMMA, async global->LDS).
//
// Roofline: 58 GFLOP, ~82MB min HBM traffic (3.5us @ 23.3TB/s). fp32 WMMA
// would be ~70-90us; single bf16 ~12us but ~0.4% error; we use a 3-term bf16
// hi/lo split (Ahi*Whi + Alo*Whi + Ahi*Wlo) -> near-fp32 accuracy at ~35us of
// v_wmma_f32_16x16x32_bf16 work.
//
// Data movement: tiles are staged with GLOBAL_LOAD_ASYNC_TO_LDS_B128
// (ASYNCcnt-tracked), double-buffered in LDS so tile k+1 streams in while
// tile k is consumed by WMMAs. K1 uses 96KB LDS, K2 64KB (of 320KB/WGP).
// ---------------------------------------------------------------------------

static constexpr int BN_  = 8192;   // batch
static constexpr int HN_  = 1024;   // hidden
static constexpr int IN_  = 128;    // input
static constexpr int KN_  = 1152;   // I + H (GEMM K)
static constexpr int TMX_ = 128;
static constexpr int NT_  = KN_ / 32;  // 36 K-tiles

typedef __attribute__((ext_vector_type(16))) __bf16 v16bf;
typedef __attribute__((ext_vector_type(8)))  __bf16 bf8_t;
typedef __attribute__((ext_vector_type(8)))  float  v8f;

union FragAB { v16bf v; bf8_t h[2]; };

__device__ __forceinline__ unsigned short f32_to_bf16_rne(float f) {
  unsigned u = __float_as_uint(f);
  u += 0x7FFFu + ((u >> 16) & 1u);
  return (unsigned short)(u >> 16);
}
__device__ __forceinline__ float bf16_to_f32(unsigned short h) {
  return __uint_as_float((unsigned)h << 16);
}
__device__ __forceinline__ void split_hilo(float f, unsigned short& hi, unsigned short& lo) {
  hi = f32_to_bf16_rne(f);
  lo = f32_to_bf16_rne(f - bf16_to_f32(hi));
}
__device__ __forceinline__ float sigmoid_f(float x) {
  return 1.0f / (1.0f + __expf(-x));
}
__device__ __forceinline__ float tanh_fast(float x) {
  float e = __expf(-2.0f * fabsf(x));
  float t = (1.0f - e) / (1.0f + e);
  return copysignf(t, x);
}
__device__ __forceinline__ v8f wmma_bf16(v16bf a, v16bf b, v8f c) {
  // v_wmma_f32_16x16x32_bf16  (neg_a, A, neg_b, B, c_mod, C, reuse_a, reuse_b)
  return __builtin_amdgcn_wmma_f32_16x16x32_bf16(false, a, false, b, (short)0, c, false, false);
}

// Async 16B copy global->LDS (per lane). VDST = 32-bit LDS byte address
// (low 32 bits of the generic pointer to __shared__), VADDR = 64-bit global.
__device__ __forceinline__ void async_cp16(const unsigned short* lds,
                                           const unsigned short* g) {
  unsigned l = (unsigned)(unsigned long long)lds;
  asm volatile("global_load_async_to_lds_b128 %0, %1, off"
               :: "v"(l), "v"((const void*)g) : "memory");
}
template <int N>
__device__ __forceinline__ void wait_asynccnt() {
  asm volatile("s_wait_asynccnt %0" :: "i"(N) : "memory");
}

// ---------------------------------------------------------------------------
// K0w: convert W_r / W_z / W_h (fp32 [H,K]) to bf16 hi/lo halves (once).
// ---------------------------------------------------------------------------
__global__ void k0_convert_weights(const float* __restrict__ Wr,
                                   const float* __restrict__ Wz,
                                   const float* __restrict__ Wh,
                                   unsigned short* __restrict__ Whi,
                                   unsigned short* __restrict__ Wlo) {
  const int mat = blockIdx.y;
  const float* src = (mat == 0) ? Wr : (mat == 1) ? Wz : Wh;
  const size_t nW = (size_t)HN_ * KN_;
  unsigned short* dh = Whi + (size_t)mat * nW;
  unsigned short* dl = Wlo + (size_t)mat * nW;
  size_t f4 = (size_t)blockIdx.x * 256 + threadIdx.x;   // < H*K/4
  float4 v = reinterpret_cast<const float4*>(src)[f4];
  ushort4 hi, lo;
  split_hilo(v.x, hi.x, lo.x); split_hilo(v.y, hi.y, lo.y);
  split_hilo(v.z, hi.z, lo.z); split_hilo(v.w, hi.w, lo.w);
  *reinterpret_cast<ushort4*>(&dh[f4 * 4]) = hi;
  *reinterpret_cast<ushort4*>(&dl[f4 * 4]) = lo;
}

// ---------------------------------------------------------------------------
// K0a: build A1 = [x_t | h] bf16 hi/lo ([B,1152]); seed x cols of A2.
// ---------------------------------------------------------------------------
__global__ void k0_build_A(const float* __restrict__ t_ptr,
                           const float* __restrict__ h,
                           const float* __restrict__ x_coeffs,
                           unsigned short* __restrict__ A1h,
                           unsigned short* __restrict__ A1l,
                           unsigned short* __restrict__ A2h,
                           unsigned short* __restrict__ A2l) {
  int ti = (int)t_ptr[0];
  ti = ti < 0 ? 0 : (ti > TMX_ - 1 ? TMX_ - 1 : ti);
  const float* x = x_coeffs + (size_t)ti * BN_ * IN_;

  size_t f4  = (size_t)blockIdx.x * 256 + threadIdx.x;  // < B*K/4
  size_t row = f4 / (KN_ / 4);
  int    c4  = (int)(f4 % (KN_ / 4));
  int    col = c4 * 4;

  float4 v;
  if (col < IN_) v = reinterpret_cast<const float4*>(x)[row * (IN_ / 4) + c4];
  else           v = reinterpret_cast<const float4*>(h)[row * (HN_ / 4) + (c4 - IN_ / 4)];

  ushort4 hi, lo;
  split_hilo(v.x, hi.x, lo.x); split_hilo(v.y, hi.y, lo.y);
  split_hilo(v.z, hi.z, lo.z); split_hilo(v.w, hi.w, lo.w);
  *reinterpret_cast<ushort4*>(&A1h[f4 * 4]) = hi;
  *reinterpret_cast<ushort4*>(&A1l[f4 * 4]) = lo;
  if (col < IN_) {
    *reinterpret_cast<ushort4*>(&A2h[f4 * 4]) = hi;
    *reinterpret_cast<ushort4*>(&A2l[f4 * 4]) = lo;
  }
}

// ---------------------------------------------------------------------------
// K1: r,z GEMMs. Block tile 128x128, 8 wave32s (wave tile 32x64). Double-
// buffered LDS filled by async global->LDS b128 copies (12 per thread per
// tile). Epilogue: z -> zbuf fp32, r*h -> A2 cols 128.. (bf16 hi/lo).
// ---------------------------------------------------------------------------
__launch_bounds__(256, 1)
__global__ void k1_rz(const unsigned short* __restrict__ A1h,
                      const unsigned short* __restrict__ A1l,
                      const unsigned short* __restrict__ Wrh,
                      const unsigned short* __restrict__ Wrl,
                      const unsigned short* __restrict__ Wzh,
                      const unsigned short* __restrict__ Wzl,
                      const float* __restrict__ h,
                      float* __restrict__ zbuf,
                      unsigned short* __restrict__ A2h,
                      unsigned short* __restrict__ A2l) {
  __shared__ __align__(16) unsigned short sAh[2][128 * 32], sAl[2][128 * 32];
  __shared__ __align__(16) unsigned short sRh[2][128 * 32], sRl[2][128 * 32];
  __shared__ __align__(16) unsigned short sZh[2][128 * 32], sZl[2][128 * 32];

  const int tid  = threadIdx.x;
  const int lane = tid & 31;
  const int wid  = tid >> 5;
  const int wm   = wid >> 1;        // 0..3 (M)
  const int wn   = wid & 1;         // 0..1 (N)
  const int l16  = lane & 15;
  const int lh   = lane >> 4;       // half-wave selector
  const int bm0  = blockIdx.y * 128;
  const int cn0  = blockIdx.x * 128;

  // issue one K-tile's worth of async copies into buffer `buf`
  auto issue_tile = [&](int k0, int buf) {
    #pragma unroll
    for (int i = 0; i < 2; ++i) {
      int c = tid + i * 256;                 // 0..511 (16B chunks)
      int r = c >> 2;
      int q = c & 3;
      size_t gA = (size_t)(bm0 + r) * KN_ + k0 + q * 8;
      size_t gW = (size_t)(cn0 + r) * KN_ + k0 + q * 8;
      async_cp16(&sAh[buf][c * 8], &A1h[gA]);
      async_cp16(&sAl[buf][c * 8], &A1l[gA]);
      async_cp16(&sRh[buf][c * 8], &Wrh[gW]);
      async_cp16(&sRl[buf][c * 8], &Wrl[gW]);
      async_cp16(&sZh[buf][c * 8], &Wzh[gW]);
      async_cp16(&sZl[buf][c * 8], &Wzl[gW]);
    }
  };

  v8f zero;
  #pragma unroll
  for (int i = 0; i < 8; ++i) zero[i] = 0.0f;
  v8f ar[2][4], az[2][4];
  #pragma unroll
  for (int tm = 0; tm < 2; ++tm)
    #pragma unroll
    for (int tn = 0; tn < 4; ++tn) { ar[tm][tn] = zero; az[tm][tn] = zero; }

  issue_tile(0, 0);

  for (int kt = 0; kt < NT_; ++kt) {
    const int cur = kt & 1;
    if (kt + 1 < NT_) {
      issue_tile((kt + 1) * 32, cur ^ 1);
      wait_asynccnt<12>();   // in-order: current tile's 12 copies are done
    } else {
      wait_asynccnt<0>();
    }
    __syncthreads();         // current buffer visible to all waves

    // A fragments: 16-bit 16x32 layout -> lanes 0-15: K0-7 & 16-23,
    // lanes 16-31: K8-15 & 24-31 (two ds_load_b128 each)
    FragAB ah[2], al[2];
    #pragma unroll
    for (int tm = 0; tm < 2; ++tm) {
      int ao = (wm * 32 + tm * 16 + l16) * 32 + lh * 8;
      ah[tm].h[0] = *reinterpret_cast<const bf8_t*>(&sAh[cur][ao]);
      ah[tm].h[1] = *reinterpret_cast<const bf8_t*>(&sAh[cur][ao + 16]);
      al[tm].h[0] = *reinterpret_cast<const bf8_t*>(&sAl[cur][ao]);
      al[tm].h[1] = *reinterpret_cast<const bf8_t*>(&sAl[cur][ao + 16]);
    }
    #pragma unroll
    for (int tn = 0; tn < 4; ++tn) {
      // B fragments: 32x16 layout -> lane = N, lanes 0-15 hold K0-7 & 16-23,
      // lanes 16-31 hold K8-15 & 24-31 (same K order as A)
      int bo = (wn * 64 + tn * 16 + l16) * 32 + lh * 8;
      FragAB brh, brl, bzh, bzl;
      brh.h[0] = *reinterpret_cast<const bf8_t*>(&sRh[cur][bo]);
      brh.h[1] = *reinterpret_cast<const bf8_t*>(&sRh[cur][bo + 16]);
      brl.h[0] = *reinterpret_cast<const bf8_t*>(&sRl[cur][bo]);
      brl.h[1] = *reinterpret_cast<const bf8_t*>(&sRl[cur][bo + 16]);
      bzh.h[0] = *reinterpret_cast<const bf8_t*>(&sZh[cur][bo]);
      bzh.h[1] = *reinterpret_cast<const bf8_t*>(&sZh[cur][bo + 16]);
      bzl.h[0] = *reinterpret_cast<const bf8_t*>(&sZl[cur][bo]);
      bzl.h[1] = *reinterpret_cast<const bf8_t*>(&sZl[cur][bo + 16]);
      #pragma unroll
      for (int tm = 0; tm < 2; ++tm) {
        ar[tm][tn] = wmma_bf16(ah[tm].v, brl.v, ar[tm][tn]);  // hi*lo
        ar[tm][tn] = wmma_bf16(al[tm].v, brh.v, ar[tm][tn]);  // lo*hi
        ar[tm][tn] = wmma_bf16(ah[tm].v, brh.v, ar[tm][tn]);  // hi*hi
        az[tm][tn] = wmma_bf16(ah[tm].v, bzl.v, az[tm][tn]);
        az[tm][tn] = wmma_bf16(al[tm].v, bzh.v, az[tm][tn]);
        az[tm][tn] = wmma_bf16(ah[tm].v, bzh.v, az[tm][tn]);
      }
    }
    __syncthreads();         // readers done before next async overwrite
  }

  // epilogue: C layout is M = vgpr + 8*(lane>=16), N = lane&15
  #pragma unroll
  for (int tm = 0; tm < 2; ++tm)
    #pragma unroll
    for (int tn = 0; tn < 4; ++tn)
      #pragma unroll
      for (int v = 0; v < 8; ++v) {
        int row = bm0 + wm * 32 + tm * 16 + lh * 8 + v;
        int col = cn0 + wn * 64 + tn * 16 + l16;
        size_t oH = (size_t)row * HN_ + col;
        float rv = sigmoid_f(ar[tm][tn][v]);
        float zv = sigmoid_f(az[tm][tn][v]);
        zbuf[oH] = zv;
        float rh = rv * h[oH];
        unsigned short hi, lo;
        split_hilo(rh, hi, lo);
        size_t oA = (size_t)row * KN_ + IN_ + col;
        A2h[oA] = hi;
        A2l[oA] = lo;
      }
}

// ---------------------------------------------------------------------------
// K2: h~ GEMM over A2 = [x | r*h], fused dh = (1-z)*(tanh(.) - h).
// Same async double-buffered pipeline (8 copies per thread per tile).
// ---------------------------------------------------------------------------
__launch_bounds__(256, 1)
__global__ void k2_h(const unsigned short* __restrict__ A2h,
                     const unsigned short* __restrict__ A2l,
                     const unsigned short* __restrict__ Whh,
                     const unsigned short* __restrict__ Whl,
                     const float* __restrict__ h,
                     const float* __restrict__ zbuf,
                     float* __restrict__ out) {
  __shared__ __align__(16) unsigned short sAh[2][128 * 32], sAl[2][128 * 32];
  __shared__ __align__(16) unsigned short sWh[2][128 * 32], sWl[2][128 * 32];

  const int tid  = threadIdx.x;
  const int lane = tid & 31;
  const int wid  = tid >> 5;
  const int wm   = wid >> 1;
  const int wn   = wid & 1;
  const int l16  = lane & 15;
  const int lh   = lane >> 4;
  const int bm0  = blockIdx.y * 128;
  const int cn0  = blockIdx.x * 128;

  auto issue_tile = [&](int k0, int buf) {
    #pragma unroll
    for (int i = 0; i < 2; ++i) {
      int c = tid + i * 256;
      int r = c >> 2;
      int q = c & 3;
      size_t gA = (size_t)(bm0 + r) * KN_ + k0 + q * 8;
      size_t gW = (size_t)(cn0 + r) * KN_ + k0 + q * 8;
      async_cp16(&sAh[buf][c * 8], &A2h[gA]);
      async_cp16(&sAl[buf][c * 8], &A2l[gA]);
      async_cp16(&sWh[buf][c * 8], &Whh[gW]);
      async_cp16(&sWl[buf][c * 8], &Whl[gW]);
    }
  };

  v8f zero;
  #pragma unroll
  for (int i = 0; i < 8; ++i) zero[i] = 0.0f;
  v8f acc[2][4];
  #pragma unroll
  for (int tm = 0; tm < 2; ++tm)
    #pragma unroll
    for (int tn = 0; tn < 4; ++tn) acc[tm][tn] = zero;

  issue_tile(0, 0);

  for (int kt = 0; kt < NT_; ++kt) {
    const int cur = kt & 1;
    if (kt + 1 < NT_) {
      issue_tile((kt + 1) * 32, cur ^ 1);
      wait_asynccnt<8>();
    } else {
      wait_asynccnt<0>();
    }
    __syncthreads();

    FragAB ah[2], al[2];
    #pragma unroll
    for (int tm = 0; tm < 2; ++tm) {
      int ao = (wm * 32 + tm * 16 + l16) * 32 + lh * 8;
      ah[tm].h[0] = *reinterpret_cast<const bf8_t*>(&sAh[cur][ao]);
      ah[tm].h[1] = *reinterpret_cast<const bf8_t*>(&sAh[cur][ao + 16]);
      al[tm].h[0] = *reinterpret_cast<const bf8_t*>(&sAl[cur][ao]);
      al[tm].h[1] = *reinterpret_cast<const bf8_t*>(&sAl[cur][ao + 16]);
    }
    #pragma unroll
    for (int tn = 0; tn < 4; ++tn) {
      int bo = (wn * 64 + tn * 16 + l16) * 32 + lh * 8;
      FragAB bh, bl;
      bh.h[0] = *reinterpret_cast<const bf8_t*>(&sWh[cur][bo]);
      bh.h[1] = *reinterpret_cast<const bf8_t*>(&sWh[cur][bo + 16]);
      bl.h[0] = *reinterpret_cast<const bf8_t*>(&sWl[cur][bo]);
      bl.h[1] = *reinterpret_cast<const bf8_t*>(&sWl[cur][bo + 16]);
      #pragma unroll
      for (int tm = 0; tm < 2; ++tm) {
        acc[tm][tn] = wmma_bf16(ah[tm].v, bl.v, acc[tm][tn]);
        acc[tm][tn] = wmma_bf16(al[tm].v, bh.v, acc[tm][tn]);
        acc[tm][tn] = wmma_bf16(ah[tm].v, bh.v, acc[tm][tn]);
      }
    }
    __syncthreads();
  }

  #pragma unroll
  for (int tm = 0; tm < 2; ++tm)
    #pragma unroll
    for (int tn = 0; tn < 4; ++tn)
      #pragma unroll
      for (int v = 0; v < 8; ++v) {
        int row = bm0 + wm * 32 + tm * 16 + lh * 8 + v;
        int col = cn0 + wn * 64 + tn * 16 + l16;
        size_t oH = (size_t)row * HN_ + col;
        float ht = tanh_fast(acc[tm][tn][v]);
        float zv = zbuf[oH];
        float hv = h[oH];
        out[oH] = (1.0f - zv) * (ht - hv);
      }
}

// ---------------------------------------------------------------------------
extern "C" void kernel_launch(void* const* d_in, const int* in_sizes, int n_in,
                              void* d_out, int out_size, void* d_ws, size_t ws_size,
                              hipStream_t stream) {
  const float* t  = (const float*)d_in[0];
  const float* h  = (const float*)d_in[1];
  const float* xc = (const float*)d_in[2];
  const float* Wr = (const float*)d_in[3];
  const float* Wz = (const float*)d_in[4];
  const float* Wh = (const float*)d_in[5];
  float* out = (float*)d_out;

  const size_t nA = (size_t)BN_ * KN_;   // 9,437,184 elems
  const size_t nW = (size_t)HN_ * KN_;   // 1,179,648 elems per matrix
  // ws layout (~123 MB): A1 hi/lo, A2 hi/lo (bf16), 3x W hi/lo (bf16), z (f32)
  char* p = (char*)d_ws;
  unsigned short* A1h = (unsigned short*)p; p += nA * 2;
  unsigned short* A1l = (unsigned short*)p; p += nA * 2;
  unsigned short* A2h = (unsigned short*)p; p += nA * 2;
  unsigned short* A2l = (unsigned short*)p; p += nA * 2;
  unsigned short* Whi = (unsigned short*)p; p += 3 * nW * 2;
  unsigned short* Wlo = (unsigned short*)p; p += 3 * nW * 2;
  float*          zb  = (float*)p;          p += (size_t)BN_ * HN_ * 4;

  k0_convert_weights<<<dim3((unsigned)(nW / 4 / 256), 3), 256, 0, stream>>>(
      Wr, Wz, Wh, Whi, Wlo);
  k0_build_A<<<(unsigned)(nA / 4 / 256), 256, 0, stream>>>(
      t, h, xc, A1h, A1l, A2h, A2l);
  k1_rz<<<dim3(HN_ / 128, BN_ / 128), 256, 0, stream>>>(
      A1h, A1l, Whi, Wlo, Whi + nW, Wlo + nW, h, zb, A2h, A2l);
  k2_h<<<dim3(HN_ / 128, BN_ / 128), 256, 0, stream>>>(
      A2h, A2l, Whi + 2 * nW, Wlo + 2 * nW, h, zb, out);
}
